// GNN_14285061226567
// MI455X (gfx1250) — compile-verified, matrix operand order checked by reading, not run on hardware
//
#include <hip/hip_runtime.h>

#define Nn   20000
#define Ee   640000
#define Ll   4
#define Hh   128
#define GG   50
#define NGg  64
#define KPE  224   // K=200 gaussian features padded to 7*32

typedef __bf16 bf16_t;
typedef __attribute__((ext_vector_type(16))) __bf16 v16bf;
typedef __attribute__((ext_vector_type(8)))  float  v8f;

__device__ __forceinline__ float lrelu(float v){ return v > 0.f ? v : 0.01f * v; }

// Load a 16x32 bf16 operand fragment from LDS (row-major, `stride` elements per row)
// per CDNA5 16-bit A/B layout: lanes 0-15 rows 0-15 K[0..7]+K[16..23], lanes 16-31 K[8..15]+K[24..31].
__device__ __forceinline__ v16bf load_frag(const bf16_t* base, int stride){
  int lane = threadIdx.x & 31;
  const bf16_t* p = base + (lane & 15) * stride + ((lane >> 4) << 3);
  union { uint4 q[2]; v16bf v; } f;
  f.q[0] = *(const uint4*)p;
  f.q[1] = *(const uint4*)(p + 16);
  return f.v;
}

// One K=32 step: each wave multiplies its 16x32 A tile against eight 32x16 B tiles (full 128 cols).
// All B fragments are loaded before any WMMA issues so LDS latency is covered once, and the
// eight v_wmma_f32_16x16x32_bf16 can issue back-to-back (distinct accumulators -> no hazard NOPs).
__device__ __forceinline__ void wave_gemm_step(const bf16_t* aL, const bf16_t* bL, v8f* c){
  int wave = threadIdx.x >> 5;
  v16bf a = load_frag(aL + wave * 16 * 32, 32);
  v16bf b[8];
  #pragma unroll
  for (int nt = 0; nt < 8; nt++) b[nt] = load_frag(bL + nt * 16 * 32, 32);
  #pragma unroll
  for (int nt = 0; nt < 8; nt++)
    c[nt] = __builtin_amdgcn_wmma_f32_16x16x32_bf16(false, a, false, b[nt], (short)0, c[nt], false, false);
}

// ---------------- prep kernels ----------------

// dst[n*Kpad+k] = (k<K) ? src[k*N+n] : 0   (weight transpose + f32->bf16)
__global__ void k_transpose_bf16(const float* __restrict__ src, bf16_t* __restrict__ dst,
                                 int K, int Nc, int Kpad){
  int i = blockIdx.x * blockDim.x + threadIdx.x;
  if (i >= Nc * Kpad) return;
  int n = i / Kpad, k = i - n * Kpad;
  float v = (k < K) ? src[(size_t)k * Nc + n] : 0.f;
  dst[i] = (bf16_t)v;
}

__global__ void k_init_ident(float* sc, float* sh){
  int i = threadIdx.x;
  if (i < Hh){ sc[i] = 1.f; sh[i] = 0.f; }
}

__global__ void k_bn_finalize(const float* __restrict__ sum, const float* __restrict__ sq,
                              const float* __restrict__ g, const float* __restrict__ be,
                              float count, float* __restrict__ scale, float* __restrict__ shift){
  int i = threadIdx.x;
  if (i >= Hh) return;
  float m  = sum[i] / count;
  float v  = sq[i] / count - m * m;
  float sc = g[i] * rsqrtf(v + 1e-5f);
  scale[i] = sc;
  shift[i] = be[i] - m * sc;
}

// h = x @ node_w + node_b   (tiny: 0.16 GFLOP, scalar is fine)
__global__ void k_node_init(const float* __restrict__ x, const float* __restrict__ w,
                            const float* __restrict__ b, float* __restrict__ h,
                            bf16_t* __restrict__ hbf){
  int i = blockIdx.x * blockDim.x + threadIdx.x;
  if (i >= Nn * Hh) return;
  int n = i >> 7, c = i & 127;
  float acc = b[c];
  #pragma unroll
  for (int k = 0; k < 32; k++) acc += x[n * 32 + k] * w[k * Hh + c];
  h[i] = acc;
  hbf[i] = (bf16_t)acc;
}

// ---------------- edge init: gaussian expand + [E,224]@[224,128] WMMA ----------------

__global__ __launch_bounds__(256) void k_edge_init(const float* __restrict__ nd,
                                                   const bf16_t* __restrict__ wt,  // [128][224] bf16
                                                   const float* __restrict__ eb,
                                                   bf16_t* __restrict__ e0bf,
                                                   float* __restrict__ decay){
  __shared__ __align__(16) bf16_t aL[128 * 32];
  __shared__ __align__(16) bf16_t bL[128 * 32];
  __shared__ float ndL[128 * 4];
  int  tid  = threadIdx.x;
  long row0 = (long)blockIdx.x * 128;
  for (int i = tid; i < 512; i += 256) ndL[i] = nd[row0 * 4 + i];
  __syncthreads();
  v8f c[8];
  #pragma unroll
  for (int j = 0; j < 8; j++) c[j] = (v8f){0,0,0,0,0,0,0,0};
  for (int ks = 0; ks < 7; ks++){
    for (int i = tid; i < 4096; i += 256){           // generate gaussian A chunk
      int r = i >> 5, k = (i & 31) + ks * 32;
      float v = 0.f;
      if (k < 200){
        int f = k / GG, g = k - f * GG;
        float d = ndL[r * 4 + f] - (float)g * (1.f / 49.f);
        v = __expf(-d * d * 1200.5f);                // (49^2)/2 = 1200.5
      }
      aL[i] = (bf16_t)v;
    }
    for (int i = tid; i < 512; i += 256){            // stage B chunk
      int n = i >> 2, q = i & 3;
      ((uint4*)bL)[i] = *(const uint4*)(wt + (size_t)n * KPE + ks * 32 + q * 8);
    }
    __syncthreads();
    wave_gemm_step(aL, bL, c);
    __syncthreads();
  }
  int lane = tid & 31, wave = tid >> 5;
  int rbase = wave * 16 + ((lane >> 4) << 3);
  int colL  = lane & 15;
  #pragma unroll
  for (int r = 0; r < 8; r++){
    long row = row0 + rbase + r;
    #pragma unroll
    for (int nt = 0; nt < 8; nt++){
      int col = nt * 16 + colL;
      e0bf[row * Hh + col] = (bf16_t)(c[nt][r] + eb[col]);
    }
  }
  if (tid < 128) decay[row0 + tid] = __cosf(1.57079632679f * ndL[tid * 4 + 3]);
}

// ---------------- edge MLP stage 1: gathered 3-block GEMM + lrelu + BN stats ----------------

__global__ __launch_bounds__(256) void k_edge_mlp1(const int* __restrict__ src, const int* __restrict__ dst,
                                                   const bf16_t* __restrict__ hbf, const bf16_t* __restrict__ e0bf,
                                                   const bf16_t* __restrict__ w1a, const bf16_t* __restrict__ w1b,
                                                   const bf16_t* __restrict__ w1c, const float* __restrict__ bias,
                                                   float* __restrict__ out,
                                                   float* __restrict__ s_sum, float* __restrict__ s_sq){
  __shared__ __align__(16) bf16_t aL[128 * 32];
  __shared__ __align__(16) bf16_t bL[128 * 32];
  __shared__ int   idxL[128];
  __shared__ float statL[256];
  int  tid  = threadIdx.x;
  long row0 = (long)blockIdx.x * 128;
  statL[tid] = 0.f;
  v8f c[8];
  #pragma unroll
  for (int j = 0; j < 8; j++) c[j] = (v8f){0,0,0,0,0,0,0,0};
  const bf16_t* wts[3] = { w1a, w1b, w1c };
  for (int ph = 0; ph < 3; ph++){
    __syncthreads();
    if (tid < 128){
      if      (ph == 0) idxL[tid] = src[row0 + tid];
      else if (ph == 2) idxL[tid] = dst[row0 + tid];
      else              idxL[tid] = (int)(row0 + tid);
    }
    __syncthreads();
    const bf16_t* Asrc = (ph == 1) ? e0bf : hbf;
    const bf16_t* W    = wts[ph];
    for (int ks = 0; ks < 4; ks++){
      for (int i = tid; i < 512; i += 256){          // gathered A chunk (16B per lane-chunk)
        int r = i >> 2, q = i & 3;
        ((uint4*)aL)[i] = *(const uint4*)(Asrc + (size_t)idxL[r] * Hh + ks * 32 + q * 8);
      }
      for (int i = tid; i < 512; i += 256){
        int n = i >> 2, q = i & 3;
        ((uint4*)bL)[i] = *(const uint4*)(W + (size_t)n * Hh + ks * 32 + q * 8);
      }
      __syncthreads();
      wave_gemm_step(aL, bL, c);
      __syncthreads();
    }
  }
  int lane = tid & 31, wave = tid >> 5;
  int rbase = wave * 16 + ((lane >> 4) << 3);
  int colL  = lane & 15;
  #pragma unroll
  for (int nt = 0; nt < 8; nt++){
    int col = nt * 16 + colL;
    float ps = 0.f, pq = 0.f;
    #pragma unroll
    for (int r = 0; r < 8; r++){
      long row = row0 + rbase + r;
      float a = lrelu(c[nt][r] + bias[col]);
      out[row * Hh + col] = a;
      ps += a; pq += a * a;
    }
    atomicAdd(&statL[col], ps);
    atomicAdd(&statL[128 + col], pq);
  }
  __syncthreads();
  if (tid < 128){
    atomicAdd(&s_sum[tid], statL[tid]);
    atomicAdd(&s_sq[tid],  statL[128 + tid]);
  }
}

// ---------------- generic 128-col GEMM with fused BN-affine A and mode-specific epilogue ------
// MODE 0: out = lrelu(A'@W+b), accumulate per-column sum/sumsq   (edge MLP2, node upd1)
// MODE 1: out = A'@W+b + addv, also emit bf16                    (node upd2, residual h)
// MODE 2: scatter: atomicAdd(uacc[dst], decay*z*msg[src])        (edge MLP3 + messages)
// MODE 3: out = z; uacc = imp*z                                  (node message GEMM, A is bf16)

struct MlpArgs {
  const float*  Af;  const bf16_t* Abf;
  const float*  in_scale; const float* in_shift;
  const bf16_t* wt;  const float* bias;
  long R;
  float* out; float* s_sum; float* s_sq;
  const float* addv; bf16_t* out_bf;
  const int* gsrc; const int* gdst; const float* decay;
  const float* msg; float* uacc; const float* imp;
};

template<int MODE>
__global__ __launch_bounds__(256) void k_mlp(MlpArgs p){
  __shared__ __align__(16) bf16_t aL[128 * 32];
  __shared__ __align__(16) bf16_t bL[128 * 32];
  __shared__ float statL[256];
  int  tid  = threadIdx.x;
  long row0 = (long)blockIdx.x * 128;
  if (MODE == 0) statL[tid] = 0.f;
  v8f c[8];
  #pragma unroll
  for (int j = 0; j < 8; j++) c[j] = (v8f){0,0,0,0,0,0,0,0};
  for (int ks = 0; ks < 4; ks++){
    if (p.Abf){
      for (int i = tid; i < 4096; i += 256){
        int r = i >> 5, k = (i & 31) + ks * 32;
        long row = row0 + r;
        float v = (row < p.R) ? (float)p.Abf[row * Hh + k] : 0.f;
        aL[i] = (bf16_t)v;
      }
    } else {
      for (int i = tid; i < 4096; i += 256){
        int r = i >> 5, k = (i & 31) + ks * 32;
        long row = row0 + r;
        float v = 0.f;
        if (row < p.R) v = p.Af[row * Hh + k] * p.in_scale[k] + p.in_shift[k];
        aL[i] = (bf16_t)v;
      }
    }
    for (int i = tid; i < 512; i += 256){
      int n = i >> 2, q = i & 3;
      ((uint4*)bL)[i] = *(const uint4*)(p.wt + (size_t)n * Hh + ks * 32 + q * 8);
    }
    __syncthreads();
    wave_gemm_step(aL, bL, c);
    __syncthreads();
  }
  int lane = tid & 31, wave = tid >> 5;
  int rbase = wave * 16 + ((lane >> 4) << 3);
  int colL  = lane & 15;

  if constexpr (MODE == 0){
    #pragma unroll
    for (int nt = 0; nt < 8; nt++){
      int col = nt * 16 + colL;
      float ps = 0.f, pq = 0.f;
      #pragma unroll
      for (int r = 0; r < 8; r++){
        long row = row0 + rbase + r;
        if (row < p.R){
          float a = lrelu(c[nt][r] + p.bias[col]);
          p.out[row * Hh + col] = a;
          ps += a; pq += a * a;
        }
      }
      atomicAdd(&statL[col], ps);
      atomicAdd(&statL[128 + col], pq);
    }
    __syncthreads();
    if (tid < 128){
      atomicAdd(&p.s_sum[tid], statL[tid]);
      atomicAdd(&p.s_sq[tid],  statL[128 + tid]);
    }
  } else if constexpr (MODE == 1){
    #pragma unroll
    for (int r = 0; r < 8; r++){
      long row = row0 + rbase + r;
      if (row < p.R){
        #pragma unroll
        for (int nt = 0; nt < 8; nt++){
          int col = nt * 16 + colL;
          float z = c[nt][r] + p.bias[col] + p.addv[row * Hh + col];
          p.out[row * Hh + col]    = z;
          p.out_bf[row * Hh + col] = (bf16_t)z;
        }
      }
    }
  } else if constexpr (MODE == 2){
    #pragma unroll
    for (int r = 0; r < 8; r++){
      long row = row0 + rbase + r;
      if (row < p.R){
        float dc = p.decay[row];
        long so = (long)p.gsrc[row] * Hh;
        long dofs = (long)p.gdst[row] * Hh;
        #pragma unroll
        for (int nt = 0; nt < 8; nt++){
          int col = nt * 16 + colL;
          float z = c[nt][r] + p.bias[col];
          atomicAdd(&p.uacc[dofs + col], dc * z * p.msg[so + col]);
        }
      }
    }
  } else {  // MODE 3
    #pragma unroll
    for (int r = 0; r < 8; r++){
      long row = row0 + rbase + r;
      if (row < p.R){
        #pragma unroll
        for (int nt = 0; nt < 8; nt++){
          int col = nt * 16 + colL;
          float z = c[nt][r] + p.bias[col];
          p.out[row * Hh + col]  = z;
          p.uacc[row * Hh + col] = p.imp[col] * z;
        }
      }
    }
  }
}

// ---------------- readout ----------------

__global__ void k_graph_accum(const float* __restrict__ h, const int* __restrict__ bidx,
                              float* __restrict__ gsum, float* __restrict__ gcnt){
  int i = blockIdx.x * blockDim.x + threadIdx.x;
  if (i >= Nn * Hh) return;
  int n = i >> 7, c = i & 127;
  int b = bidx[n];
  atomicAdd(&gsum[b * Hh + c], h[i]);
  if (c == 0) atomicAdd(&gcnt[b], 1.f);
}

__global__ __launch_bounds__(256) void k_readout(const float* __restrict__ gsum, const float* __restrict__ gcnt,
                                                 const float* __restrict__ w1, const float* __restrict__ b1,
                                                 const float* __restrict__ g,  const float* __restrict__ be,
                                                 const float* __restrict__ w2, const float* __restrict__ b2,
                                                 float* __restrict__ out){
  __shared__ float gm[NGg * Hh];
  __shared__ float z1[NGg * 64];
  __shared__ float af[128];
  int tid = threadIdx.x;
  for (int i = tid; i < NGg * Hh; i += 256){
    int gr = i >> 7;
    float cnt = gcnt[gr]; if (cnt < 1.f) cnt = 1.f;
    gm[i] = gsum[i] / cnt;
  }
  __syncthreads();
  for (int i = tid; i < NGg * 64; i += 256){
    int gr = i >> 6, c = i & 63;
    float acc = b1[c];
    for (int k = 0; k < Hh; k++) acc += gm[gr * Hh + k] * w1[k * 64 + c];
    z1[i] = lrelu(acc);
  }
  __syncthreads();
  if (tid < 64){
    float s = 0.f, ss = 0.f;
    for (int gr = 0; gr < NGg; gr++){ float v = z1[gr * 64 + tid]; s += v; ss += v * v; }
    float m = s / NGg, var = ss / NGg - m * m;
    float sc = g[tid] * rsqrtf(var + 1e-5f);
    af[tid] = sc; af[64 + tid] = be[tid] - m * sc;
  }
  __syncthreads();
  if (tid < NGg){
    float acc = b2[0];
    for (int c = 0; c < 64; c++) acc += (z1[tid * 64 + c] * af[c] + af[64 + c]) * w2[c];
    out[tid] = acc * 2.0f + (-5.0f);   // * E_STD + E_MEAN
  }
}

// ---------------- host driver ----------------

extern "C" void kernel_launch(void* const* d_in, const int* in_sizes, int n_in,
                              void* d_out, int out_size, void* d_ws, size_t ws_size,
                              hipStream_t stream){
  (void)in_sizes; (void)n_in; (void)out_size; (void)ws_size;
  const float* x      = (const float*)d_in[0];
  const int*   ei     = (const int*)  d_in[1];
  const int*   src    = ei;
  const int*   dst    = ei + Ee;
  const float* nd     = (const float*)d_in[2];
  const int*   bidx   = (const int*)  d_in[3];
  const float* edge_w = (const float*)d_in[4];
  const float* edge_b = (const float*)d_in[5];
  const float* node_w = (const float*)d_in[6];
  const float* node_b = (const float*)d_in[7];
  const float* int_w1 = (const float*)d_in[8];
  const float* int_b1 = (const float*)d_in[9];
  const float* int_g1 = (const float*)d_in[10];
  const float* int_be1= (const float*)d_in[11];
  const float* int_w2 = (const float*)d_in[12];
  const float* int_b2 = (const float*)d_in[13];
  const float* int_g2 = (const float*)d_in[14];
  const float* int_be2= (const float*)d_in[15];
  const float* int_w3 = (const float*)d_in[16];
  const float* int_b3 = (const float*)d_in[17];
  const float* mes_w  = (const float*)d_in[18];
  const float* mes_b  = (const float*)d_in[19];
  const float* imp    = (const float*)d_in[20];
  const float* upd_w1 = (const float*)d_in[21];
  const float* upd_b1 = (const float*)d_in[22];
  const float* upd_g  = (const float*)d_in[23];
  const float* upd_be = (const float*)d_in[24];
  const float* upd_w2 = (const float*)d_in[25];
  const float* upd_b2 = (const float*)d_in[26];
  const float* tgt_w1 = (const float*)d_in[27];
  const float* tgt_b1 = (const float*)d_in[28];
  const float* tgt_g  = (const float*)d_in[29];
  const float* tgt_be = (const float*)d_in[30];
  const float* tgt_w2 = (const float*)d_in[31];
  const float* tgt_b2 = (const float*)d_in[32];

  char* base = (char*)d_ws;
  size_t off = 0;
  auto alloc = [&](size_t bytes) -> void* {
    off = (off + 255) & ~(size_t)255;
    void* r = base + off;
    off += bytes;
    return r;
  };
  float*  h    = (float*) alloc((size_t)Nn * Hh * 4);
  bf16_t* hbf  = (bf16_t*)alloc((size_t)Nn * Hh * 2);
  bf16_t* e0bf = (bf16_t*)alloc((size_t)Ee * Hh * 2);
  float*  t    = (float*) alloc((size_t)Ee * Hh * 4);
  float*  dec  = (float*) alloc((size_t)Ee * 4);
  float*  msg  = (float*) alloc((size_t)Nn * Hh * 4);
  float*  uacc = (float*) alloc((size_t)Nn * Hh * 4);
  float*  u1   = (float*) alloc((size_t)Nn * Hh * 4);
  bf16_t* wtE  = (bf16_t*)alloc((size_t)Hh * KPE * 2);
  bf16_t* wtL  = (bf16_t*)alloc((size_t)Ll * 8 * Hh * Hh * 2);
  float*  stats= (float*) alloc(6 * Hh * 4);      // sum1,sq1,sum2,sq2,sumU,sqU
  float*  aff  = (float*) alloc(8 * Hh * 4);      // sc1,sh1,sc2,sh2,scU,shU,id_sc,id_sh
  float*  gsum = (float*) alloc((size_t)NGg * Hh * 4);
  float*  gcnt = (float*) alloc((size_t)NGg * 4);

  // ---- weight prep (tiny) ----
  k_transpose_bf16<<<(Hh * KPE + 255) / 256, 256, 0, stream>>>(edge_w, wtE, 200, Hh, KPE);
  for (int l = 0; l < Ll; l++){
    const float* srcs[8] = {
      int_w1 + (size_t)l * 384 * Hh,
      int_w1 + (size_t)l * 384 * Hh + (size_t)128 * Hh,
      int_w1 + (size_t)l * 384 * Hh + (size_t)256 * Hh,
      int_w2 + (size_t)l * Hh * Hh,
      int_w3 + (size_t)l * Hh * Hh,
      mes_w  + (size_t)l * Hh * Hh,
      upd_w1 + (size_t)l * Hh * Hh,
      upd_w2 + (size_t)l * Hh * Hh };
    for (int j = 0; j < 8; j++)
      k_transpose_bf16<<<(Hh * Hh + 255) / 256, 256, 0, stream>>>(
          srcs[j], wtL + (size_t)(l * 8 + j) * Hh * Hh, Hh, Hh, Hh);
  }
  k_init_ident<<<1, 128, 0, stream>>>(aff + 6 * Hh, aff + 7 * Hh);

  // ---- node + edge init ----
  k_node_init<<<(Nn * Hh + 255) / 256, 256, 0, stream>>>(x, node_w, node_b, h, hbf);
  k_edge_init<<<Ee / 128, 256, 0, stream>>>(nd, wtE, edge_b, e0bf, dec);

  const int EB = Ee / 128;                 // 5000
  const int NB = (Nn + 127) / 128;         // 157

  for (int l = 0; l < Ll; l++){
    bf16_t* w1a = wtL + (size_t)(l * 8 + 0) * Hh * Hh;
    bf16_t* w1b = wtL + (size_t)(l * 8 + 1) * Hh * Hh;
    bf16_t* w1c = wtL + (size_t)(l * 8 + 2) * Hh * Hh;
    bf16_t* w2  = wtL + (size_t)(l * 8 + 3) * Hh * Hh;
    bf16_t* w3  = wtL + (size_t)(l * 8 + 4) * Hh * Hh;
    bf16_t* wm  = wtL + (size_t)(l * 8 + 5) * Hh * Hh;
    bf16_t* wu1 = wtL + (size_t)(l * 8 + 6) * Hh * Hh;
    bf16_t* wu2 = wtL + (size_t)(l * 8 + 7) * Hh * Hh;

    hipMemsetAsync(stats, 0, 6 * Hh * 4, stream);

    // t = lrelu(cat@W1 + b1), stats1
    k_edge_mlp1<<<EB, 256, 0, stream>>>(src, dst, hbf, e0bf, w1a, w1b, w1c,
                                        int_b1 + l * Hh, t, stats, stats + Hh);
    k_bn_finalize<<<1, 128, 0, stream>>>(stats, stats + Hh, int_g1 + l * Hh, int_be1 + l * Hh,
                                         (float)Ee, aff, aff + Hh);
    // t = lrelu(BN1(t)@W2 + b2) in place, stats2
    {
      MlpArgs a{}; a.Af = t; a.in_scale = aff; a.in_shift = aff + Hh;
      a.wt = w2; a.bias = int_b2 + l * Hh; a.R = Ee;
      a.out = t; a.s_sum = stats + 2 * Hh; a.s_sq = stats + 3 * Hh;
      k_mlp<0><<<EB, 256, 0, stream>>>(a);
    }
    k_bn_finalize<<<1, 128, 0, stream>>>(stats + 2 * Hh, stats + 3 * Hh, int_g2 + l * Hh,
                                         int_be2 + l * Hh, (float)Ee, aff + 2 * Hh, aff + 3 * Hh);
    // msg = h@mes_w + b; uacc = imp*msg
    {
      MlpArgs a{}; a.Abf = hbf; a.wt = wm; a.bias = mes_b + l * Hh; a.R = Nn;
      a.out = msg; a.uacc = uacc; a.imp = imp + l * Hh;
      k_mlp<3><<<NB, 256, 0, stream>>>(a);
    }
    // z = BN2(t)@W3 + b3; scatter uacc[dst] += decay*z*msg[src]
    {
      MlpArgs a{}; a.Af = t; a.in_scale = aff + 2 * Hh; a.in_shift = aff + 3 * Hh;
      a.wt = w3; a.bias = int_b3 + l * Hh; a.R = Ee;
      a.gsrc = src; a.gdst = dst; a.decay = dec; a.msg = msg; a.uacc = uacc;
      k_mlp<2><<<EB, 256, 0, stream>>>(a);
    }
    // u1 = lrelu(uacc@upd_w1 + b), statsU
    {
      MlpArgs a{}; a.Af = uacc; a.in_scale = aff + 6 * Hh; a.in_shift = aff + 7 * Hh;
      a.wt = wu1; a.bias = upd_b1 + l * Hh; a.R = Nn;
      a.out = u1; a.s_sum = stats + 4 * Hh; a.s_sq = stats + 5 * Hh;
      k_mlp<0><<<NB, 256, 0, stream>>>(a);
    }
    k_bn_finalize<<<1, 128, 0, stream>>>(stats + 4 * Hh, stats + 5 * Hh, upd_g + l * Hh,
                                         upd_be + l * Hh, (float)Nn, aff + 4 * Hh, aff + 5 * Hh);
    // h = BNu(u1)@upd_w2 + b + h  (residual), refresh hbf
    {
      MlpArgs a{}; a.Af = u1; a.in_scale = aff + 4 * Hh; a.in_shift = aff + 5 * Hh;
      a.wt = wu2; a.bias = upd_b2 + l * Hh; a.R = Nn;
      a.out = h; a.addv = h; a.out_bf = hbf;
      k_mlp<1><<<NB, 256, 0, stream>>>(a);
    }
  }

  // ---- readout ----
  hipMemsetAsync(gsum, 0, (size_t)NGg * Hh * 4, stream);
  hipMemsetAsync(gcnt, 0, (size_t)NGg * 4, stream);
  k_graph_accum<<<(Nn * Hh + 255) / 256, 256, 0, stream>>>(h, bidx, gsum, gcnt);
  k_readout<<<1, 256, 0, stream>>>(gsum, gcnt, tgt_w1, tgt_b1, tgt_g, tgt_be,
                                   tgt_w2, tgt_b2, (float*)d_out);
}
